// STCNet_74028056314291
// MI455X (gfx1250) — matrix operand order repaired and checked by reading, not compiled, not run on hardware
//
#include <hip/hip_runtime.h>
#include <hip/hip_bf16.h>

// ---------------------------------------------------------------------------
// STCNet (criss-cross attention + window attention) for gfx1250 (MI455X).
// All heavy GEMMs go through v_wmma_f32_16x16x32_bf16 (wave32 WMMA).
// Data layouts chosen so fragment gathers are unit-stride in K (mergeable to
// b128 loads); activation tiles staged via LDS; ASYNCcnt-tracked
// global->LDS staging via global_load_async_to_lds_b128 when available.
// ---------------------------------------------------------------------------

typedef __attribute__((ext_vector_type(16))) __bf16 v16bf;
typedef __attribute__((ext_vector_type(8)))  float  v8f;

constexpr int   kB   = 8;
constexpr int   kC   = 256;
constexpr int   kH   = 128;
constexpr int   kW   = 128;
constexpr int   kHW  = kH * kW;       // 16384
constexpr float kNEG = -1000000000.0f;

#if defined(__has_builtin)
#if __has_builtin(__builtin_amdgcn_global_load_async_to_lds_b128) && \
    __has_builtin(__builtin_amdgcn_s_wait_asynccnt)
#define STC_ASYNC_LDS 1
#endif
#endif
#ifndef STC_ASYNC_LDS
#define STC_ASYNC_LDS 0
#endif

#if STC_ASYNC_LDS
// builtin signature (from hipcc diagnostic): param0 = int __vector(4) AS1*,
// i.e. global source; param1 = LDS destination; then imm offset, imm cpol.
typedef int v4i_a __attribute__((vector_size(16)));
typedef __attribute__((address_space(1))) v4i_a* gas_ptr;
typedef __attribute__((address_space(3))) v4i_a* las_ptr;
#endif

// ---- WMMA helpers ----------------------------------------------------------

__device__ inline v8f wmma_bf16(v16bf a, v16bf b, v8f c) {
  // D = A(16x32 bf16) * B(32x16 bf16) + C(16x16 f32)
  return __builtin_amdgcn_wmma_f32_16x16x32_bf16(
      /*neg_a=*/false, a, /*neg_b=*/false, b,
      /*c_mod=*/(short)0, c, /*reuse_a=*/false, /*reuse_b=*/false);
}

// A fragment: 16x32 (M x K) bf16. lane l in [0,15]: M=l; [16,31]: M=l-16.
// half j: K = (j&7) + 16*(j>>3) + 8*(l>>4)   (per ISA 16-bit A layout)
template <typename T>
__device__ inline v16bf frag_a(const T* p, int sm, int sk, int lane) {
  const int g = (lane >> 4) & 1, m = lane & 15;
  v16bf a;
#pragma unroll
  for (int j = 0; j < 16; ++j) {
    const int k = (j & 7) + ((j >> 3) << 4) + 8 * g;
    a[j] = (__bf16)p[(long)m * sm + (long)k * sk];
  }
  return a;
}

// B fragment: 32x16 (K x N) bf16. lane l: N = l&15; half j: K = (j&7) + 16*(j>>3) + 8*(l>>4)
template <typename T>
__device__ inline v16bf frag_b(const T* p, int sk, int sn, int lane) {
  const int g = (lane >> 4) & 1, n = lane & 15;
  v16bf b;
#pragma unroll
  for (int j = 0; j < 16; ++j) {
    const int k = (j & 7) + ((j >> 3) << 4) + 8 * g;
    b[j] = (__bf16)p[(long)k * sk + (long)n * sn];
  }
  return b;
}

// C/D fragment mapping: vgpr r, lane l -> M = r + 8*(l>>4), N = l&15.

// ---------------------------------------------------------------------------
// Kernel 1: q/k projections.  q = Wq(32x256) @ x[b] + bq (k likewise).
// Output PIXEL-MAJOR: q[b][hw][32] so later energy gathers are unit-stride.
// grid (HW/64, B), block 128 (4 waves): wave 0/1 -> q m-tiles, 2/3 -> k.
// x tile staged in LDS once per block (waves share it 4x).
// ---------------------------------------------------------------------------
__global__ void k_qk_proj(const float* __restrict__ x,
                          const float* __restrict__ Wq, const float* __restrict__ bq,
                          const float* __restrict__ Wk, const float* __restrict__ bk,
                          float* __restrict__ qo, float* __restrict__ ko) {
  __shared__ __bf16 xsT[64][36];  // [pixel][c-chunk], transposed, padded
  const int lane = threadIdx.x & 31, wid = threadIdx.x >> 5;
  const int b = blockIdx.y;
  const int pix0 = blockIdx.x * 64;
  const bool is_k = wid >= 2;
  const int m0 = (wid & 1) * 16;
  const float* Wm = is_k ? Wk : Wq;
  const float* bm = is_k ? bk : bq;
  float* outp = is_k ? ko : qo;
  const float* xb = x + (long)b * kC * kHW;

  v8f acc[4] = {};
  for (int k0 = 0; k0 < kC; k0 += 32) {
    __syncthreads();  // protect previous iteration's reads
    // stage x[k0..k0+32][pix0..pix0+64] -> xsT (coalesced float4 loads)
    for (int e = threadIdx.x; e < 512; e += 128) {
      const int row = e >> 4, col4 = (e & 15) * 4;
      const float4 v4 = *reinterpret_cast<const float4*>(
          xb + (long)(k0 + row) * kHW + pix0 + col4);
      xsT[col4 + 0][row] = (__bf16)v4.x;
      xsT[col4 + 1][row] = (__bf16)v4.y;
      xsT[col4 + 2][row] = (__bf16)v4.z;
      xsT[col4 + 3][row] = (__bf16)v4.w;
    }
    __syncthreads();
    v16bf a = frag_a(Wm + (long)m0 * kC + k0, kC, 1, lane);  // unit-stride K
#pragma unroll
    for (int nt = 0; nt < 4; ++nt) {
      v16bf bb = frag_b(&xsT[nt * 16][0], 1, 36, lane);      // unit-stride K
      acc[nt] = wmma_bf16(a, bb, acc[nt]);
    }
  }
  const int g = (lane >> 4) & 1, n = lane & 15;
#pragma unroll
  for (int nt = 0; nt < 4; ++nt)
#pragma unroll
    for (int r = 0; r < 8; ++r) {
      const int m = m0 + r + 8 * g;
      const int pix = pix0 + nt * 16 + n;
      outp[(long)b * kHW * 32 + (long)pix * 32 + m] = acc[nt][r] + bm[m];
    }
}

// ---------------------------------------------------------------------------
// Kernel 2: v projection. v[b,c,hw] = Wv(256x256) @ y[b] + bv  (BCHW kept)
// grid (HW/64, C/64, B), block 128 (4 waves); y tile staged in LDS (4x reuse).
// ---------------------------------------------------------------------------
__global__ void k_v_proj(const float* __restrict__ y,
                         const float* __restrict__ Wv, const float* __restrict__ bv,
                         float* __restrict__ vo) {
  __shared__ __bf16 ysT[64][36];
  const int lane = threadIdx.x & 31, wid = threadIdx.x >> 5;
  const int b = blockIdx.z;
  const int pix0 = blockIdx.x * 64;
  const int m0 = blockIdx.y * 64 + wid * 16;
  const float* yb = y + (long)b * kC * kHW;

  v8f acc[4] = {};
  for (int k0 = 0; k0 < kC; k0 += 32) {
    __syncthreads();
    for (int e = threadIdx.x; e < 512; e += 128) {
      const int row = e >> 4, col4 = (e & 15) * 4;
      const float4 v4 = *reinterpret_cast<const float4*>(
          yb + (long)(k0 + row) * kHW + pix0 + col4);
      ysT[col4 + 0][row] = (__bf16)v4.x;
      ysT[col4 + 1][row] = (__bf16)v4.y;
      ysT[col4 + 2][row] = (__bf16)v4.z;
      ysT[col4 + 3][row] = (__bf16)v4.w;
    }
    __syncthreads();
    v16bf a = frag_a(Wv + (long)m0 * kC + k0, kC, 1, lane);
#pragma unroll
    for (int nt = 0; nt < 4; ++nt) {
      v16bf bb = frag_b(&ysT[nt * 16][0], 1, 36, lane);
      acc[nt] = wmma_bf16(a, bb, acc[nt]);
    }
  }
  const int g = (lane >> 4) & 1, n = lane & 15;
#pragma unroll
  for (int nt = 0; nt < 4; ++nt)
#pragma unroll
    for (int r = 0; r < 8; ++r) {
      const int m = m0 + r + 8 * g;
      vo[(long)b * kC * kHW + (long)m * kHW + pix0 + nt * 16 + n] =
          acc[nt][r] + bv[m];
    }
}

// ---------------------------------------------------------------------------
// Kernel 3: criss-cross energies (q/k are pixel-major [b][hw][32]).
// mode 0: eH[b,h,idx,g] = sum_c q[b,(h,idx),c]*k[b,(g,idx),c] + diag(NEG)
// mode 1: eW[b,idx,w,v] = sum_c q[b,(idx,w),c]*k[b,(idx,v),c]
// grid (128, B, 2), block 256 (8 waves x 8 tiles). K = 32 -> single WMMA.
// All gathers unit-stride in K -> b128-mergeable.
// ---------------------------------------------------------------------------
__global__ void k_energy(const float* __restrict__ q, const float* __restrict__ k,
                         float* __restrict__ eH, float* __restrict__ eW) {
  const int lane = threadIdx.x & 31, wid = threadIdx.x >> 5;
  const int idx = blockIdx.x;
  const int b = blockIdx.y;
  const int mode = blockIdx.z;
  const float* qb = q + (long)b * kHW * 32;
  const float* kb = k + (long)b * kHW * 32;

  for (int i = 0; i < 8; ++i) {
    const int t = wid * 8 + i;            // 64 tiles of 16x16
    const int m0 = (t >> 3) * 16, n0 = (t & 7) * 16;
    v16bf a, bb;
    if (mode == 0) {  // fixed w=idx; pixel (h,idx) stride = kW*32
      a  = frag_a(qb + ((long)m0 * kW + idx) * 32, kW * 32, 1, lane);
      bb = frag_b(kb + ((long)n0 * kW + idx) * 32, 1, kW * 32, lane);
    } else {          // fixed h=idx; pixel (idx,w) stride = 32
      a  = frag_a(qb + ((long)idx * kW + m0) * 32, 32, 1, lane);
      bb = frag_b(kb + ((long)idx * kW + n0) * 32, 1, 32, lane);
    }
    v8f acc = {};
    acc = wmma_bf16(a, bb, acc);
    const int g = (lane >> 4) & 1, n = lane & 15;
    if (mode == 0) {
      float* p = eH + ((long)b * kH * kW + idx) * kH;  // [b][h][idx][g]
#pragma unroll
      for (int r = 0; r < 8; ++r) {
        const int m = m0 + r + 8 * g, nn = n0 + n;
        p[(long)m * (kW * kH) + nn] = acc[r] + (m == nn ? kNEG : 0.0f);
      }
    } else {
      float* p = eW + (((long)b * kH + idx) * kW) * kW;  // [b][idx][w][v]
#pragma unroll
      for (int r = 0; r < 8; ++r) {
        const int m = m0 + r + 8 * g, nn = n0 + n;
        p[(long)m * kW + nn] = acc[r];
      }
    }
  }
}

// ---------------------------------------------------------------------------
// Kernel 4: joint softmax over [eH_row(128) ++ eW_row(128)] per pixel.
// One wave (32 lanes) per pixel, 8 pixels per block. In-place -> aH, aW.
// ---------------------------------------------------------------------------
__global__ void k_softmax(float* __restrict__ eH, float* __restrict__ eW) {
  const int lane = threadIdx.x & 31, wid = threadIdx.x >> 5;
  const long p = (long)blockIdx.x * 8 + wid;   // flat (b,h,w)
  float* rH = eH + p * kH;
  float* rW = eW + p * kW;
  float v[8];
#pragma unroll
  for (int i = 0; i < 4; ++i) v[i]     = rH[lane + 32 * i];
#pragma unroll
  for (int i = 0; i < 4; ++i) v[4 + i] = rW[lane + 32 * i];
  float mx = v[0];
#pragma unroll
  for (int i = 1; i < 8; ++i) mx = fmaxf(mx, v[i]);
  for (int o = 16; o; o >>= 1) mx = fmaxf(mx, __shfl_xor(mx, o, 32));
  float s = 0.0f;
#pragma unroll
  for (int i = 0; i < 8; ++i) { v[i] = __expf(v[i] - mx); s += v[i]; }
  for (int o = 16; o; o >>= 1) s += __shfl_xor(s, o, 32);
  const float inv = 1.0f / s;
#pragma unroll
  for (int i = 0; i < 4; ++i) rH[lane + 32 * i] = v[i] * inv;
#pragma unroll
  for (int i = 0; i < 4; ++i) rW[lane + 32 * i] = v[4 + i] * inv;
}

// ---------------------------------------------------------------------------
// Kernel 5a: oH (fixed b,w): oH[c,h] = sum_g v[c,g,w] * aH[h,g]
// Writes cc[b,h,w,c] = gamma*oH + x[b,c,h,w]   (BHWC for window stage)
// grid (W, B), block 256. A-fragments hoisted across the 8 h-tiles (8x reuse).
// ---------------------------------------------------------------------------
__global__ void k_oH(const float* __restrict__ v, const float* __restrict__ aH,
                     const float* __restrict__ x, const float* __restrict__ gamma,
                     float* __restrict__ cc) {
  const int lane = threadIdx.x & 31, wid = threadIdx.x >> 5;
  const int w = blockIdx.x, b = blockIdx.y;
  const float* vb = v + (long)b * kC * kHW;
  const float gm = gamma[0];

  for (int ct = 0; ct < 2; ++ct) {
    const int c0 = (wid * 2 + ct) * 16;
    v16bf afr[4];
#pragma unroll
    for (int s = 0; s < 4; ++s)
      afr[s] = frag_a(vb + (long)c0 * kHW + (long)(s * 32) * kW + w, kHW, kW, lane);
    for (int ht = 0; ht < 8; ++ht) {
      const int h0 = ht * 16;
      v8f acc = {};
#pragma unroll
      for (int s = 0; s < 4; ++s) {
        v16bf bb = frag_b(aH + (((long)b * kH + h0) * kW + w) * kH + s * 32,
                          1, kW * kH, lane);  // unit-stride K
        acc = wmma_bf16(afr[s], bb, acc);
      }
      const int g = (lane >> 4) & 1, n = lane & 15;
#pragma unroll
      for (int r = 0; r < 8; ++r) {
        const int c = c0 + r + 8 * g, h = h0 + n;
        cc[(((long)b * kH + h) * kW + w) * kC + c] =
            gm * acc[r] + x[(long)b * kC * kHW + (long)c * kHW + (long)h * kW + w];
      }
    }
  }
}

// ---------------------------------------------------------------------------
// Kernel 5b: oW (fixed b,h): oW[c,w] = sum_v v[c,h,v] * aW[w,v]
// cc[b,h,w,c] += gamma*oW     (runs after k_oH on the same stream)
// ---------------------------------------------------------------------------
__global__ void k_oW(const float* __restrict__ v, const float* __restrict__ aW,
                     const float* __restrict__ gamma, float* __restrict__ cc) {
  const int lane = threadIdx.x & 31, wid = threadIdx.x >> 5;
  const int h = blockIdx.x, b = blockIdx.y;
  const float* vb = v + (long)b * kC * kHW;
  const float gm = gamma[0];

  for (int ct = 0; ct < 2; ++ct) {
    const int c0 = (wid * 2 + ct) * 16;
    v16bf afr[4];
#pragma unroll
    for (int s = 0; s < 4; ++s)
      afr[s] = frag_a(vb + (long)c0 * kHW + (long)h * kW + s * 32, kHW, 1, lane);
    for (int wt = 0; wt < 8; ++wt) {
      const int w0 = wt * 16;
      v8f acc = {};
#pragma unroll
      for (int s = 0; s < 4; ++s) {
        v16bf bb = frag_b(aW + (((long)b * kH + h) * kW + w0) * kW + s * 32,
                          1, kW, lane);  // unit-stride K
        acc = wmma_bf16(afr[s], bb, acc);
      }
      const int g = (lane >> 4) & 1, n = lane & 15;
#pragma unroll
      for (int r = 0; r < 8; ++r) {
        const int c = c0 + r + 8 * g, w = w0 + n;
        const long o = (((long)b * kH + h) * kW + w) * kC + c;
        cc[o] += gm * acc[r];
      }
    }
  }
}

// ---------------------------------------------------------------------------
// Kernel 6: CPB MLP -> 16*sigmoid(rpb[head][64][64]); grid(8)=heads.
// ---------------------------------------------------------------------------
__device__ inline float relcoord(int d) {
  const float tv = (float)d / 7.0f * 8.0f;
  const float sg = (tv > 0.0f) ? 1.0f : ((tv < 0.0f) ? -1.0f : 0.0f);
  return sg * __log2f(fabsf(tv) + 1.0f) / 3.0f;  // log2(8)=3
}

__global__ void k_rpb(const float* __restrict__ w1, const float* __restrict__ b1,
                      const float* __restrict__ w2, float* __restrict__ rpb) {
  __shared__ float tbl[225];
  const int head = blockIdx.x, t = threadIdx.x;
  for (int ti = t; ti < 225; ti += blockDim.x) {
    const int dh = ti / 15 - 7, dw = ti % 15 - 7;
    const float c0 = relcoord(dh), c1 = relcoord(dw);
    float s = 0.0f;
    for (int u = 0; u < 512; ++u) {
      const float hv = fmaxf(c0 * w1[2 * u] + c1 * w1[2 * u + 1] + b1[u], 0.0f);
      s += hv * w2[head * 512 + u];
    }
    tbl[ti] = s;
  }
  __syncthreads();
  for (int p = t; p < 4096; p += blockDim.x) {
    const int i = p >> 6, j = p & 63;
    const int dh = (i >> 3) - (j >> 3) + 7, dw = (i & 7) - (j & 7) + 7;
    rpb[head * 4096 + p] = 16.0f / (1.0f + __expf(-tbl[dh * 15 + dw]));
  }
}

// ---------------------------------------------------------------------------
// Kernel 7: fused window attention. One workgroup (8 waves) per 8x8 window.
// Dynamic LDS: X/O bf16 [64][260], QKV bf16 [768][68], ATT f32 [64][66],
// plus (async path) fp32 staging tile [64][256].
// ---------------------------------------------------------------------------
constexpr int XS_S  = 260;
constexpr int QKV_S = 68;
constexpr int ATT_S = 66;
constexpr size_t kQKVOff = (size_t)64 * XS_S * 2;             // 33280
constexpr size_t kATTOff = kQKVOff + (size_t)768 * QKV_S * 2; // 137728
constexpr size_t kCCFOff = kATTOff + (size_t)64 * ATT_S * 4;  // 154624 (16B aligned)
constexpr size_t kWinSmem = kCCFOff + (STC_ASYNC_LDS ? (size_t)64 * 256 * 4 : 0);

__global__ void k_winattn(const float* __restrict__ cc, const float* __restrict__ qkv_w,
                          const float* __restrict__ q_bias, const float* __restrict__ v_bias,
                          const float* __restrict__ logit_scale, const float* __restrict__ rpb,
                          const float* __restrict__ proj_w, const float* __restrict__ proj_b,
                          float* __restrict__ out) {
  extern __shared__ char smem[];
  __bf16* XS  = (__bf16*)smem;                 // 64 x 260 (X, later O)
  __bf16* QKV = (__bf16*)(smem + kQKVOff);     // 768 x 68
  float*  ATT = (float*)(smem + kATTOff);      // 64 x 66

  const int tid = threadIdx.x, lane = tid & 31, wid = tid >> 5;
  const int win = blockIdx.x;
  const int b = win >> 8, wh = (win >> 4) & 15, ww = win & 15;

  // --- phase 1: stage window tokens (64 x 256) into LDS ---------------------
#if STC_ASYNC_LDS
  {
    float* CCF = (float*)(smem + kCCFOff);
    for (int i = 0; i < 16; ++i) {             // 4096 x 16B async chunks
      const int e = tid + i * 256;
      const int n = e >> 6, c4 = (e & 63) * 4;
      const int h = wh * 8 + (n >> 3), w = ww * 8 + (n & 7);
      const float* gp = cc + (((long)b * kH + h) * kW + w) * kC + c4;
      __builtin_amdgcn_global_load_async_to_lds_b128(
          (gas_ptr)gp, (las_ptr)(CCF + n * 256 + c4), 0, 0);
    }
    __builtin_amdgcn_s_wait_asynccnt(0);
    __syncthreads();
    for (int i = 0; i < 64; ++i) {
      const int e = tid + i * 256;
      XS[(e >> 8) * XS_S + (e & 255)] = (__bf16)CCF[e];
    }
  }
#else
  for (int n = 0; n < 64; ++n) {
    const int h = wh * 8 + (n >> 3), w = ww * 8 + (n & 7);
    XS[n * XS_S + tid] = (__bf16)cc[(((long)b * kH + h) * kW + w) * kC + tid];
  }
#endif
  if (wid == 0) {  // warm L2 for the shared qkv weights (global_prefetch_b8)
    for (int o = lane; o < 768; o += 32) __builtin_prefetch(qkv_w + (long)o * kC, 0, 0);
  }
  __syncthreads();

  // --- phase 2: qkv = X @ qkv_w^T + bias   (M=64, N=768, K=256) ------------
  for (int i = 0; i < 24; ++i) {
    const int t = wid * 24 + i;               // 192 tiles
    const int m0 = (t / 48) * 16, o0 = (t % 48) * 16;
    v8f acc = {};
    for (int k0 = 0; k0 < kC; k0 += 32) {
      v16bf a  = frag_a(XS + m0 * XS_S + k0, XS_S, 1, lane);
      v16bf bb = frag_b(qkv_w + (long)o0 * kC + k0, 1, kC, lane);
      acc = wmma_bf16(a, bb, acc);
    }
    const int g = (lane >> 4) & 1, nn = lane & 15;
#pragma unroll
    for (int r = 0; r < 8; ++r) {
      const int tok = m0 + r + 8 * g, o = o0 + nn;
      const float bias = (o < 256) ? q_bias[o] : ((o < 512) ? 0.0f : v_bias[o - 512]);
      QKV[o * QKV_S + tok] = (__bf16)(acc[r] + bias);
    }
  }
  __syncthreads();

  // --- phase 3: cosine-normalize q (x scale) and k per (head, token) -------
  for (int it = 0; it < 4; ++it) {
    const int item = tid * 4 + it;            // 1024 = 2(sel) x 8(head) x 64(tok)
    const int sel = item >> 9, head = (item >> 6) & 7, tok = item & 63;
    __bf16* row = QKV + (sel * 256 + head * 32) * QKV_S + tok;
    float ss = 0.0f;
    for (int d = 0; d < 32; ++d) { const float vv = (float)row[d * QKV_S]; ss += vv * vv; }
    float inv = 1.0f / fmaxf(sqrtf(ss), 1e-12f);
    if (sel == 0)
      inv *= __expf(fminf(logit_scale[head], 4.605170185988092f));  // ln(100)
    for (int d = 0; d < 32; ++d)
      row[d * QKV_S] = (__bf16)((float)row[d * QKV_S] * inv);
  }
  __syncthreads();

  // --- phase 4: per-head attention -----------------------------------------
  for (int head = 0; head < 8; ++head) {
    const __bf16* Q  = QKV + (head * 32) * QKV_S;
    const __bf16* Kp = QKV + (256 + head * 32) * QKV_S;
    const __bf16* Vp = QKV + (512 + head * 32) * QKV_S;

    // logits = qn @ kn^T (scale folded into qn) + 16*sigmoid(rpb); 64x64, K=32
    for (int i = 0; i < 2; ++i) {
      const int t = wid * 2 + i;              // 16 tiles
      const int m0 = (t >> 2) * 16, n0 = (t & 3) * 16;
      v16bf a  = frag_a(Q + m0, 1, QKV_S, lane);        // A(m=tok,k=d)
      v16bf bb = frag_b(Kp + n0, QKV_S, 1, lane);       // B(k=d,n=tok2)
      v8f acc = {};
      acc = wmma_bf16(a, bb, acc);
      const int g = (lane >> 4) & 1, nn = lane & 15;
#pragma unroll
      for (int r = 0; r < 8; ++r) {
        const int m = m0 + r + 8 * g, n2 = n0 + nn;
        ATT[m * ATT_S + n2] = acc[r] + rpb[head * 4096 + m * 64 + n2];
      }
    }
    __syncthreads();

    // row softmax (wave per row, 2 cols per lane)
    for (int i = 0; i < 8; ++i) {
      const int m = wid * 8 + i;
      const float v0 = ATT[m * ATT_S + lane];
      const float v1 = ATT[m * ATT_S + 32 + lane];
      float mx = fmaxf(v0, v1);
      for (int o = 16; o; o >>= 1) mx = fmaxf(mx, __shfl_xor(mx, o, 32));
      const float e0 = __expf(v0 - mx), e1 = __expf(v1 - mx);
      float s = e0 + e1;
      for (int o = 16; o; o >>= 1) s += __shfl_xor(s, o, 32);
      const float inv = 1.0f / s;
      ATT[m * ATT_S + lane]      = e0 * inv;
      ATT[m * ATT_S + 32 + lane] = e1 * inv;
    }
    __syncthreads();

    // O[:, head*32 : head*32+32] = attn @ V    (64x32, K=64) -> reuse XS
    {
      const int m0 = (wid >> 1) * 16, d0 = (wid & 1) * 16;   // 8 tiles, 1/wave
      v8f acc = {};
      for (int k0 = 0; k0 < 64; k0 += 32) {
        v16bf a  = frag_a(ATT + m0 * ATT_S + k0, ATT_S, 1, lane);
        v16bf bb = frag_b(Vp + d0 * QKV_S + k0, 1, QKV_S, lane);
        acc = wmma_bf16(a, bb, acc);
      }
      const int g = (lane >> 4) & 1, nn = lane & 15;
#pragma unroll
      for (int r = 0; r < 8; ++r) {
        const int tok = m0 + r + 8 * g, d = d0 + nn;
        XS[tok * XS_S + head * 32 + d] = (__bf16)acc[r];
      }
    }
    __syncthreads();
  }

  // --- phase 5: out = O @ proj_w^T + proj_b, scatter to BCHW ---------------
  for (int i = 0; i < 8; ++i) {
    const int t = wid * 8 + i;                // 64 tiles
    const int m0 = (t >> 4) * 16, o0 = (t & 15) * 16;
    v8f acc = {};
    for (int k0 = 0; k0 < kC; k0 += 32) {
      v16bf a  = frag_a(XS + m0 * XS_S + k0, XS_S, 1, lane);
      v16bf bb = frag_b(proj_w + (long)o0 * kC + k0, 1, kC, lane);
      acc = wmma_bf16(a, bb, acc);
    }
    const int g = (lane >> 4) & 1, nn = lane & 15;
#pragma unroll
    for (int r = 0; r < 8; ++r) {
      const int tok = m0 + r + 8 * g, o = o0 + nn;
      const int h = wh * 8 + (tok >> 3), w = ww * 8 + (tok & 7);
      out[(((long)b * kC + o) * kH + h) * kW + w] = acc[r] + proj_b[o];
    }
  }
}

// ---------------------------------------------------------------------------
// Launcher
// ---------------------------------------------------------------------------
extern "C" void kernel_launch(void* const* d_in, const int* in_sizes, int n_in,
                              void* d_out, int out_size, void* d_ws, size_t ws_size,
                              hipStream_t stream) {
  (void)in_sizes; (void)n_in; (void)out_size; (void)ws_size;
  const float* x       = (const float*)d_in[0];
  const float* y       = (const float*)d_in[1];
  const float* Wq      = (const float*)d_in[2];
  const float* bq      = (const float*)d_in[3];
  const float* Wk      = (const float*)d_in[4];
  const float* bk      = (const float*)d_in[5];
  const float* Wv      = (const float*)d_in[6];
  const float* bv      = (const float*)d_in[7];
  const float* gamma   = (const float*)d_in[8];
  const float* qkv_w   = (const float*)d_in[9];
  const float* q_bias  = (const float*)d_in[10];
  const float* v_bias  = (const float*)d_in[11];
  const float* lscale  = (const float*)d_in[12];
  const float* cpb_w1  = (const float*)d_in[13];
  const float* cpb_b1  = (const float*)d_in[14];
  const float* cpb_w2  = (const float*)d_in[15];
  const float* proj_w  = (const float*)d_in[16];
  const float* proj_b  = (const float*)d_in[17];
  float* out = (float*)d_out;

  // workspace carve-up (floats)
  float* ws   = (float*)d_ws;
  const size_t qk_sz = (size_t)kB * 32 * kHW;          //  4 Mi floats each
  const size_t v_sz  = (size_t)kB * kC * kHW;          // 32 Mi
  const size_t e_sz  = (size_t)kB * kH * kW * kH;      // 16 Mi each
  const size_t cc_sz = (size_t)kB * kH * kW * kC;      // 32 Mi
  float* q_ws  = ws;
  float* k_ws  = q_ws + qk_sz;
  float* v_ws  = k_ws + qk_sz;
  float* eH_ws = v_ws + v_sz;
  float* eW_ws = eH_ws + e_sz;
  float* cc_ws = eW_ws + e_sz;
  float* rpb_ws = cc_ws + cc_sz;                       // 8*64*64 floats

  // 1) criss-cross projections
  k_qk_proj<<<dim3(kHW / 64, kB), 128, 0, stream>>>(x, Wq, bq, Wk, bk, q_ws, k_ws);
  k_v_proj <<<dim3(kHW / 64, kC / 64, kB), 128, 0, stream>>>(y, Wv, bv, v_ws);
  // 2) energies + joint softmax
  k_energy <<<dim3(kW, kB, 2), 256, 0, stream>>>(q_ws, k_ws, eH_ws, eW_ws);
  k_softmax<<<dim3((kB * kHW) / 8), 256, 0, stream>>>(eH_ws, eW_ws);
  // 3) oH/oW + residual -> cc (BHWC)
  k_oH<<<dim3(kW, kB), 256, 0, stream>>>(v_ws, eH_ws, x, gamma, cc_ws);
  k_oW<<<dim3(kH, kB), 256, 0, stream>>>(v_ws, eW_ws, gamma, cc_ws);
  // 4) relative-position bias table
  k_rpb<<<dim3(8), 256, 0, stream>>>(cpb_w1, cpb_b1, cpb_w2, rpb_ws);
  // 5) fused window attention (one WGP workgroup per 8x8 window)
  k_winattn<<<dim3(kB * (kH / 8) * (kW / 8)), 256, kWinSmem, stream>>>(
      cc_ws, qkv_w, q_bias, v_bias, lscale, rpb_ws, proj_w, proj_b, out);
}
